// AttentionLayer_12695923327322
// MI455X (gfx1250) — matrix operand order repaired and checked by reading, not compiled, not run on hardware
//
#include <hip/hip_runtime.h>
#include <hip/hip_bf16.h>

typedef __attribute__((ext_vector_type(16))) _Float16 v16h;
typedef __attribute__((ext_vector_type(8)))  _Float16 v8h;
typedef __attribute__((ext_vector_type(2)))  _Float16 v2h;
typedef __attribute__((ext_vector_type(8)))  float    v8f;

#define LSEQ   200
#define EDIM   64
#define HDIM   128
#define FDIM   256          // 4*E
#define KP     264          // row pitch (halves) for feat / W1t in LDS, mult of 8
#define NTILES 13           // ceil(200/16)

// one block per batch row; 256 threads = 8 wave32
__global__ __launch_bounds__(256) void din_attn_wmma(
    const float* __restrict__ behavior,   // [B, L, E]
    const float* __restrict__ target,     // [B, 1, E]
    const int*   __restrict__ seq_len,    // [B]
    const float* __restrict__ W1,         // [4E, H] row-major
    const float* __restrict__ b1,         // [H]
    const float* __restrict__ W2,         // [H, 1]
    const float* __restrict__ b2,         // [1]
    float* __restrict__ out)              // [B, E]
{
    extern __shared__ char smem[];
    _Float16* sW1t  = (_Float16*)smem;              // [H][KP] f16 : W1 transposed
    _Float16* sFeat = sW1t + HDIM * KP;             // 8 waves * [16][KP] f16
    float*    sAtt  = (float*)(sFeat + 8 * 16 * KP);// [208]
    float*    sRed  = sAtt + 208;                   // [256]

    const int tid  = threadIdx.x;
    const int ln   = tid & 31;
    const int wave = tid >> 5;
    const int nIdx = ln & 15;
    const int b    = blockIdx.x;
    const int slen = seq_len[b];

    // ---- stage W1^T into LDS as f16: sW1t[h][k] = W1[k*H + h] ----
    for (int idx = tid; idx < FDIM * HDIM; idx += 256) {
        int k = idx >> 7;            // 0..255
        int h = idx & (HDIM - 1);    // 0..127
        sW1t[h * KP + k] = (_Float16)W1[idx];
    }

    // per-lane N-column constants for the 8 N-tiles (N = t*16 + nIdx)
    float b1v[8], w2v[8];
#pragma unroll
    for (int t = 0; t < 8; ++t) {
        b1v[t] = b1[t * 16 + nIdx];
        w2v[t] = W2[t * 16 + nIdx];
    }
    const float b2v = b2[0];

    // per-lane target pair: e = 2*ln, 2*ln+1
    const int e0 = ln * 2;
    const float2 tv = *(const float2*)(target + (size_t)b * EDIM + e0);

    __syncthreads();   // sW1t visible to all waves

    _Float16* feat = sFeat + wave * (16 * KP);

#pragma unroll 1
    for (int it = 0; it < 2; ++it) {
        const int tile = wave + it * 8;
        if (tile < NTILES) {
            const int tileBase = tile * 16;

            // ---- build 16x256 f16 feature tile (wave-private LDS) ----
#pragma unroll
            for (int r = 0; r < 16; ++r) {
                const int l = tileBase + r;
                float2 bv = make_float2(0.f, 0.f);
                if (l < LSEQ)
                    bv = *(const float2*)(behavior + ((size_t)b * LSEQ + l) * EDIM + e0);
                v2h hb; hb[0] = (_Float16)bv.x;           hb[1] = (_Float16)bv.y;
                v2h ht; ht[0] = (_Float16)tv.x;           ht[1] = (_Float16)tv.y;
                v2h hm; hm[0] = (_Float16)(bv.x * tv.x);  hm[1] = (_Float16)(bv.y * tv.y);
                v2h hd; hd[0] = (_Float16)(bv.x - tv.x);  hd[1] = (_Float16)(bv.y - tv.y);
                _Float16* row = feat + r * KP;
                *(v2h*)(row +       e0) = hb;
                *(v2h*)(row +  64 + e0) = ht;
                *(v2h*)(row + 128 + e0) = hm;
                *(v2h*)(row + 192 + e0) = hd;
            }
            asm volatile("s_wait_dscnt 0x0" ::: "memory");

            // ---- h = feat @ W1 : 8 N-tiles x 8 K-steps of wmma f16 ----
            v8f acc[8];
            const v8f vzero = {0.f, 0.f, 0.f, 0.f, 0.f, 0.f, 0.f, 0.f};
#pragma unroll
            for (int t = 0; t < 8; ++t) acc[t] = vzero;

#pragma unroll
            for (int s = 0; s < 8; ++s) {
                // A fragment: lane holds row M=nIdx, K = base..base+15 contiguous
                const int kA = s * 32 + ((ln < 16) ? 0 : 16);
                const _Float16* ar = feat + nIdx * KP;
                v8h a0 = *(const v8h*)(ar + kA);
                v8h a1 = *(const v8h*)(ar + kA + 8);
                v16h afrag = __builtin_shufflevector(a0, a1,
                    0, 1, 2, 3, 4, 5, 6, 7, 8, 9, 10, 11, 12, 13, 14, 15);
                // B fragment: lane holds col N = t*16+nIdx, K = base..base+15 contiguous
                const int kB = s * 32 + ((ln < 16) ? 0 : 16);
#pragma unroll
                for (int t = 0; t < 8; ++t) {
                    const _Float16* br = sW1t + (t * 16 + nIdx) * KP + kB;
                    v8h bb0 = *(const v8h*)(br);
                    v8h bb1 = *(const v8h*)(br + 8);
                    v16h bfrag = __builtin_shufflevector(bb0, bb1,
                        0, 1, 2, 3, 4, 5, 6, 7, 8, 9, 10, 11, 12, 13, 14, 15);
                    acc[t] = __builtin_amdgcn_wmma_f32_16x16x32_f16(
                        false, afrag, false, bfrag, (short)0, acc[t], false, false);
                }
            }

            // ---- att[l] = relu(h + b1) . W2 + b2, masked ----
#pragma unroll
            for (int v = 0; v < 8; ++v) {
                float r = 0.f;
#pragma unroll
                for (int t = 0; t < 8; ++t) {
                    float hv = acc[t][v] + b1v[t];
                    hv = fmaxf(hv, 0.f);
                    r = fmaf(hv, w2v[t], r);
                }
                // reduce over the 16 N-lanes of each half (C layout: halves hold M=v / M=v+8)
                r += __shfl_xor(r, 1, 32);
                r += __shfl_xor(r, 2, 32);
                r += __shfl_xor(r, 4, 32);
                r += __shfl_xor(r, 8, 32);
                if (nIdx == v) {
                    const int row = v + ((ln >> 4) << 3);
                    const int l = tileBase + row;
                    float val = r + b2v;
                    if (l >= LSEQ)       val = -__builtin_inff();  // pad rows -> weight 0
                    else if (l >= slen)  val = -1e9f;              // reference mask value
                    sAtt[l] = val;
                }
            }
        }
    }
    __syncthreads();

    // ---- softmax over 208 entries (wave 0) ----
    if (wave == 0) {
        float vloc[7];
        float m = -__builtin_inff();
#pragma unroll
        for (int i = 0; i < 7; ++i) {
            const int idx = ln + 32 * i;
            float x = (idx < 208) ? sAtt[idx] : -__builtin_inff();
            vloc[i] = x;
            m = fmaxf(m, x);
        }
        m = fmaxf(m, __shfl_xor(m, 1, 32));
        m = fmaxf(m, __shfl_xor(m, 2, 32));
        m = fmaxf(m, __shfl_xor(m, 4, 32));
        m = fmaxf(m, __shfl_xor(m, 8, 32));
        m = fmaxf(m, __shfl_xor(m, 16, 32));
        float sum = 0.f;
#pragma unroll
        for (int i = 0; i < 7; ++i) {
            const int idx = ln + 32 * i;
            if (idx < 208) {
                float e = __expf(vloc[i] - m);
                sAtt[idx] = e;
                sum += e;
            }
        }
        sum += __shfl_xor(sum, 1, 32);
        sum += __shfl_xor(sum, 2, 32);
        sum += __shfl_xor(sum, 4, 32);
        sum += __shfl_xor(sum, 8, 32);
        sum += __shfl_xor(sum, 16, 32);
        const float inv = 1.0f / sum;
#pragma unroll
        for (int i = 0; i < 7; ++i) {
            const int idx = ln + 32 * i;
            if (idx < 208) sAtt[idx] *= inv;
        }
    }
    __syncthreads();

    // ---- weighted_emb[e] = sum_l att[l] * behavior[b,l,e] ----
    const int e = tid & 63;
    const int c = tid >> 6;           // 4 chunks of 50 positions
    float accw = 0.f;
    for (int l = c * 50; l < c * 50 + 50; ++l)
        accw = fmaf(sAtt[l], behavior[((size_t)b * LSEQ + l) * EDIM + e], accw);
    sRed[tid] = accw;
    __syncthreads();
    if (tid < 64)
        out[(size_t)b * EDIM + tid] =
            sRed[tid] + sRed[64 + tid] + sRed[128 + tid] + sRed[192 + tid];
}

extern "C" void kernel_launch(void* const* d_in, const int* in_sizes, int n_in,
                              void* d_out, int out_size, void* d_ws, size_t ws_size,
                              hipStream_t stream) {
    const float* behavior = (const float*)d_in[0];
    const float* target   = (const float*)d_in[1];
    const int*   seq_len  = (const int*)d_in[2];
    const float* W1       = (const float*)d_in[3];
    const float* b1       = (const float*)d_in[4];
    const float* W2       = (const float*)d_in[5];
    const float* b2       = (const float*)d_in[6];
    float* out = (float*)d_out;

    const int B = in_sizes[2];   // seq_len has one entry per batch row
    const size_t ldsBytes =
        (size_t)(HDIM * KP + 8 * 16 * KP) * sizeof(_Float16) + (208 + 256) * sizeof(float);

    din_attn_wmma<<<dim3(B), dim3(256), ldsBytes, stream>>>(
        behavior, target, seq_len, W1, b1, W2, b2, out);
}
